// DiTBlock_22728966930761
// MI455X (gfx1250) — matrix operand order repaired and checked by reading, not compiled, not run on hardware
//
#include <hip/hip_runtime.h>
#include <cmath>

// ---------------- fixed problem dimensions ----------------
constexpr int B_   = 2;
constexpr int S_   = 1024;
constexpr int LAT_ = 64;
constexpr int E_   = 1024;
constexpr int H_   = 8;
constexpr int NB_  = 4;
constexpr int HD_  = 64;     // E/H/2
constexpr int ROT_ = 32;     // HD/2
constexpr int E4_  = 4 * E_;
constexpr int M_   = B_ * S_;   // 2048 token rows

#define DEVFN __device__ __forceinline__

// ---------------- CDNA5 WMMA types ----------------
typedef __attribute__((ext_vector_type(16))) __bf16 v16bf;
typedef __attribute__((ext_vector_type(8)))  float  v8f;

union Frag16 {
  v16bf          m;
  unsigned short h[16];
  unsigned int   w[8];
  uint4          q[2];
};

DEVFN unsigned short f32_bf16(float f) {
  unsigned int u = __float_as_uint(f);
  u += 0x7FFFu + ((u >> 16) & 1u);   // round-to-nearest-even
  return (unsigned short)(u >> 16);
}

// ---- CDNA5 async global->LDS copy (ASYNCcnt-tracked DMA path) ----
DEVFN void ldsAsyncB128(void* lds, const void* g) {
  unsigned off = (unsigned)(unsigned long long)lds;   // low 32 bits of AS(3) ptr = LDS offset
  asm volatile("global_load_async_to_lds_b128 %0, %1, off"
               :: "v"(off), "v"(g) : "memory");
}
DEVFN void waitAsync0() { asm volatile("s_wait_asynccnt 0" ::: "memory"); }

DEVFN float block_sum_256(float v, float* sh) {
  int t = threadIdx.x;
  sh[t] = v; __syncthreads();
  for (int o = 128; o > 0; o >>= 1) {
    if (t < o) sh[t] += sh[t + o];
    __syncthreads();
  }
  float r = sh[0];
  __syncthreads();
  return r;
}

// ---------------- fp32 -> bf16 (row-major copy, for activations) ----------------
__global__ void k_f32_to_bf16(const float* __restrict__ src,
                              unsigned short* __restrict__ dst, int n) {
  int i = blockIdx.x * blockDim.x + threadIdx.x;
  if (i < n) dst[i] = f32_bf16(src[i]);
}

// ---------------- fp32 W[K][N] -> bf16 Wt[N][K] (tiled transpose) ----------------
__global__ __launch_bounds__(256) void k_transpose_bf16(
    const float* __restrict__ W, unsigned short* __restrict__ Wt, int K, int N)
{
  __shared__ unsigned short tile[32][33];
  const int kt = blockIdx.y * 32, nt = blockIdx.x * 32;
  const int tx = threadIdx.x & 31, ty = threadIdx.x >> 5;   // 32 x 8
  #pragma unroll
  for (int i = 0; i < 4; ++i) {
    int k = kt + ty + i * 8;
    tile[tx][ty + i * 8] = f32_bf16(W[(size_t)k * N + nt + tx]);  // coalesced in tx
  }
  __syncthreads();
  #pragma unroll
  for (int i = 0; i < 4; ++i) {
    int n = nt + ty + i * 8;
    Wt[(size_t)n * K + kt + tx] = tile[ty + i * 8][tx];           // coalesced in tx
  }
}

// ---------------- bf16 WMMA GEMM, double-buffered async LDS staging ----------------
// C[M,N] = act( A[M,K] @ Bt[N,K]^T + bias ).
// Block: 256 threads = 8 waves (2 M x 4 N). Block tile 32 x 256.
// Wave tile 16 x 64 -> 4 accumulators, 4 v_wmma per K-step.
// Bt tile (256 x 32 = 16 KB) DMA'd by global_load_async_to_lds_b128 into a
// double buffer: tile t+1's DMA overlaps tile t's WMMAs; one barrier/step.
__global__ __launch_bounds__(256) void k_gemm_bf16(
    const unsigned short* __restrict__ A,   // M x K
    const unsigned short* __restrict__ Bt,  // N x K
    const float* __restrict__ bias,
    const float* __restrict__ act_a, const float* __restrict__ act_b, // snake params
    void* __restrict__ Cout, int M, int N, int K, int act_mode, int out_bf16)
{
  __shared__ unsigned short sW[2][256][40];  // [buf][n_local][k] (+8 pad) = 40 KB

  const int tid  = threadIdx.x;
  const int lane = tid & 31, wave = tid >> 5;
  const int waveM = wave >> 2, waveN = wave & 3;
  const int mBase  = blockIdx.y * 32 + waveM * 16;
  const int nBlock = blockIdx.x * 256;
  const int lm = lane & 15, lh = lane >> 4;

  v8f acc[4];
  v8f z = {};
  #pragma unroll
  for (int f = 0; f < 4; ++f) acc[f] = z;

  const unsigned short* aRow = A + (size_t)(mBase + lm) * K;
  int nStage = nBlock + tid;
  if (nStage > N - 1) nStage = N - 1;               // clamp for N < 256 edge
  const unsigned short* bRow = Bt + (size_t)nStage * K;

  // prologue: DMA first Bt tile into buffer 0
  ldsAsyncB128(&sW[0][tid][0],  bRow + 0);
  ldsAsyncB128(&sW[0][tid][8],  bRow + 8);
  ldsAsyncB128(&sW[0][tid][16], bRow + 16);
  ldsAsyncB128(&sW[0][tid][24], bRow + 24);

  for (int k0 = 0; k0 < K; k0 += 32) {
    const int p = (k0 >> 5) & 1;
    waitAsync0();          // my group issued last iter (tile p) is complete
    __syncthreads();       // all waves' DMA landed; all reads of buf 1-p finished
    if (k0 + 32 < K) {     // kick off next tile's DMA; overlaps this tile's math
      ldsAsyncB128(&sW[1 - p][tid][0],  bRow + k0 + 32);
      ldsAsyncB128(&sW[1 - p][tid][8],  bRow + k0 + 40);
      ldsAsyncB128(&sW[1 - p][tid][16], bRow + k0 + 48);
      ldsAsyncB128(&sW[1 - p][tid][24], bRow + k0 + 56);
      __builtin_prefetch(aRow + k0 + 32, 0, 0);
    }

    // A fragment (16-bit A 16x32): h[0..15] -> K = lh*16+0..15
    Frag16 fa;
    fa.q[0] = *(const uint4*)(aRow + k0 + lh * 16);
    fa.q[1] = *(const uint4*)(aRow + k0 + lh * 16 + 8);

    // batch all four B fragments, then the four WMMAs (graduated dscnt waits)
    Frag16 fb[4];
    #pragma unroll
    for (int f = 0; f < 4; ++f) {
      const uint4* pW = (const uint4*)&sW[p][waveN * 64 + f * 16 + lm][lh * 16];
      fb[f].q[0] = pW[0]; fb[f].q[1] = pW[1];
    }
    #pragma unroll
    for (int f = 0; f < 4; ++f)
      acc[f] = __builtin_amdgcn_wmma_f32_16x16x32_bf16(false, fa.m, false, fb[f].m,
                                                       (short)0, acc[f], false, false);
  }

  float* Cf = (float*)Cout;
  unsigned short* Ch = (unsigned short*)Cout;
  #pragma unroll
  for (int f = 0; f < 4; ++f) {
    #pragma unroll
    for (int r = 0; r < 8; ++r) {
      int m = mBase + lh * 8 + r;                 // D layout: row = 8*(lane>>4)+r
      int n = nBlock + waveN * 64 + f * 16 + lm;  // col = lane&15
      if (n < N) {
        float v = acc[f][r];
        if (bias) v += bias[n];
        if (act_mode == 1) {                      // snake: x + sin(a*x)^2/(b+1e-9)
          float al = act_a[n], be = act_b[n];
          float sn = __sinf(al * v);
          v += sn * sn / (be + 1e-9f);
        }
        size_t idx = (size_t)m * N + n;
        if (out_bf16) Ch[idx] = f32_bf16(v);
        else          Cf[idx] = v;
      }
    }
  }
}

// ---------------- small M=2 matvec: mods = emb @ W + b ----------------
__global__ void k_emb_mm(const float* __restrict__ emb, const float* __restrict__ W,
                         const float* __restrict__ bias, float* __restrict__ out, int N) {
  int n = blockIdx.x * blockDim.x + threadIdx.x;
  if (n >= N) return;
  float a0 = 0.f, a1 = 0.f;
  for (int k = 0; k < E_; ++k) {
    float w = W[(size_t)k * N + n];
    a0 += emb[k] * w;
    a1 += emb[E_ + k] * w;
  }
  float bv = bias[n];
  out[n]     = a0 + bv;
  out[N + n] = a1 + bv;
}

// ---------------- modulated layernorm -> bf16 ----------------
__global__ __launch_bounds__(256) void k_mod_ln(
    const float* __restrict__ X, const float* __restrict__ w, const float* __restrict__ b,
    const float* __restrict__ mods, int modsStride, int scOff, int shOff, float eps,
    unsigned short* __restrict__ out)
{
  __shared__ float red[256];
  int row = blockIdx.x;
  int batch = row / S_;
  const float* x = X + (size_t)row * E_;
  int t = threadIdx.x;
  float s = 0.f, s2 = 0.f;
  for (int c = t; c < E_; c += 256) { float v = x[c]; s += v; s2 += v * v; }
  s  = block_sum_256(s,  red);
  s2 = block_sum_256(s2, red);
  float mean = s * (1.0f / E_);
  float var  = s2 * (1.0f / E_) - mean * mean;
  float inv  = rsqrtf(var + eps);
  const float* mrow = mods + (size_t)batch * modsStride;
  for (int c = t; c < E_; c += 256) {
    float v = (x[c] - mean) * inv;
    if (w) v = v * w[c] + b[c];
    v = v * (1.f + mrow[scOff + c]) + mrow[shOff + c];
    out[(size_t)row * E_ + c] = f32_bf16(v);
  }
}

// ---------------- rotary + repack qkv -> bf16 Q/K/Vt ----------------
// qkv f32 [M][3E]. Q/K bf16 [(b*2+map)*H+h][S][HD];  V transposed: Vt [b*H+h][128][S]
__global__ __launch_bounds__(256) void k_rotary_pack(
    const float* __restrict__ qkv, unsigned short* __restrict__ Qb,
    unsigned short* __restrict__ Kb, unsigned short* __restrict__ Vt)
{
  int row = blockIdx.x;
  int b = row / S_, sPos = row % S_;
  const float* xr = qkv + (size_t)row * (3 * E_);
  for (int c = threadIdx.x; c < 3 * E_; c += 256) {
    int sec = c >> 10;           // 0=q 1=k 2=v
    int cl  = c & (E_ - 1);
    if (sec < 2) {
      int head2 = cl >> 6;       // 0..15
      int d = cl & 63;
      const float* base = xr + sec * E_;
      float v = base[cl];
      if (d < ROT_) {
        int tt = d >> 1;
        float f = (float)sPos * __powf(10000.f, -(float)tt / 16.f);
        float cs = __cosf(f), sn = __sinf(f);
        if ((d & 1) == 0) v = v * cs - base[cl + 1] * sn;
        else              v = v * cs + base[cl - 1] * sn;
      }
      int map = head2 & 1, hh = head2 >> 1;
      size_t idx = ((size_t)((b * 2 + map) * H_ + hh) * S_ + sPos) * HD_ + d;
      (sec == 0 ? Qb : Kb)[idx] = f32_bf16(v);
    } else {
      int hh = cl >> 7, e2 = cl & 127;
      Vt[((size_t)(b * H_ + hh) * 128 + e2) * S_ + sPos] = f32_bf16(xr[2 * E_ + cl]);
    }
  }
}

// ---------------- lambda scalar ----------------
__global__ void k_lambda(const float* __restrict__ lq1, const float* __restrict__ lk1,
                         const float* __restrict__ lq2, const float* __restrict__ lk2,
                         float lam_init, float* __restrict__ lam_out) {
  __shared__ float s1[64], s2[64];
  int t = threadIdx.x;
  s1[t] = lq1[t] * lk1[t];
  s2[t] = lq2[t] * lk2[t];
  __syncthreads();
  for (int o = 32; o > 0; o >>= 1) {
    if (t < o) { s1[t] += s1[t + o]; s2[t] += s2[t + o]; }
    __syncthreads();
  }
  if (t == 0) *lam_out = __expf(s1[0]) - __expf(s2[0]) + lam_init;
}

// ---------------- flash differential attention (one map) ----------------
// grid: (S/64, H, B*2). 128 threads = 4 independent waves; 16 queries each.
// All WMMA operands are contiguous b128 global loads (V is pre-transposed);
// only the P-relayout round-trips through wave-private LDS. No barriers.
__global__ __launch_bounds__(128) void k_flash_attn(
    const unsigned short* __restrict__ Qb, const unsigned short* __restrict__ Kb,
    const unsigned short* __restrict__ Vt,
    float* __restrict__ A1, float* __restrict__ A2)
{
  __shared__ unsigned short sP[4][16][34];  // per-wave P tile 16q x 32k

  const int tid = threadIdx.x;
  const int lane = tid & 31, wave = tid >> 5;
  const int lm = lane & 15, lh = lane >> 4;
  const int qt = blockIdx.x, hh = blockIdx.y;
  const int b = blockIdx.z >> 1, map = blockIdx.z & 1;

  const unsigned short* Qh = Qb + (size_t)((b * 2 + map) * H_ + hh) * S_ * HD_;
  const unsigned short* Kh = Kb + (size_t)((b * 2 + map) * H_ + hh) * S_ * HD_;
  const unsigned short* Vh = Vt + (size_t)(b * H_ + hh) * 128 * S_;
  float* Am = map ? A2 : A1;

  const int qRow0 = qt * 64 + wave * 16;

  // Q fragments for both HD halves (A 16x32 layout, contiguous K runs)
  Frag16 fq0, fq1;
  const unsigned short* qr = Qh + (size_t)(qRow0 + lm) * HD_;
  fq0.q[0] = *(const uint4*)(qr + lh * 16);
  fq0.q[1] = *(const uint4*)(qr + lh * 16 + 8);
  fq1.q[0] = *(const uint4*)(qr + 32 + lh * 16);
  fq1.q[1] = *(const uint4*)(qr + 32 + lh * 16 + 8);

  float rowmax[8], rowsum[8];
  v8f accO[8];
  v8f vzero = {};
  #pragma unroll
  for (int r = 0; r < 8; ++r) { rowmax[r] = -1e30f; rowsum[r] = 0.f; }
  #pragma unroll
  for (int e = 0; e < 8; ++e) accO[e] = vzero;

  const float scale = 0.125f;          // HD^-0.5
  const int kend = qRow0 + 16;         // this wave's causal bound

  for (int kb = 0; kb < kend; kb += 32) {
    // scores: S[16q x 32k]; batch the 4 K-fragments, then 4 WMMAs
    Frag16 fk[4];  // [0]=keys lm,hs0 [1]=keys 16+lm,hs0 [2]=keys lm,hs1 [3]=keys 16+lm,hs1
    {
      const unsigned short* kp0 = Kh + (size_t)(kb + lm) * HD_ + lh * 16;
      const unsigned short* kp1 = Kh + (size_t)(kb + 16 + lm) * HD_ + lh * 16;
      fk[0].q[0] = ((const uint4*)kp0)[0];        fk[0].q[1] = ((const uint4*)kp0)[1];
      fk[1].q[0] = ((const uint4*)kp1)[0];        fk[1].q[1] = ((const uint4*)kp1)[1];
      fk[2].q[0] = ((const uint4*)(kp0 + 32))[0]; fk[2].q[1] = ((const uint4*)(kp0 + 32))[1];
      fk[3].q[0] = ((const uint4*)(kp1 + 32))[0]; fk[3].q[1] = ((const uint4*)(kp1 + 32))[1];
    }
    v8f s0 = {}, s1 = {};
    s0 = __builtin_amdgcn_wmma_f32_16x16x32_bf16(false, fq0.m, false, fk[0].m, (short)0, s0, false, false);
    s1 = __builtin_amdgcn_wmma_f32_16x16x32_bf16(false, fq0.m, false, fk[1].m, (short)0, s1, false, false);
    s0 = __builtin_amdgcn_wmma_f32_16x16x32_bf16(false, fq1.m, false, fk[2].m, (short)0, s0, false, false);
    s1 = __builtin_amdgcn_wmma_f32_16x16x32_bf16(false, fq1.m, false, fk[3].m, (short)0, s1, false, false);

    // online softmax per row (rows live in 16-lane half-wave groups)
    #pragma unroll
    for (int r = 0; r < 8; ++r) {
      int qi = qRow0 + lh * 8 + r;
      float v0 = (kb + lm      <= qi) ? s0[r] * scale : -1e30f;
      float v1 = (kb + 16 + lm <= qi) ? s1[r] * scale : -1e30f;
      float nm = fmaxf(v0, v1);
      nm = fmaxf(nm, __shfl_xor(nm, 1, 32));
      nm = fmaxf(nm, __shfl_xor(nm, 2, 32));
      nm = fmaxf(nm, __shfl_xor(nm, 4, 32));
      nm = fmaxf(nm, __shfl_xor(nm, 8, 32));
      nm = fmaxf(nm, rowmax[r]);
      float p0 = __expf(v0 - nm);
      float p1 = __expf(v1 - nm);
      float fs = __expf(rowmax[r] - nm);
      float ps = p0 + p1;
      ps += __shfl_xor(ps, 1, 32);
      ps += __shfl_xor(ps, 2, 32);
      ps += __shfl_xor(ps, 4, 32);
      ps += __shfl_xor(ps, 8, 32);
      rowsum[r] = rowsum[r] * fs + ps;
      rowmax[r] = nm;
      #pragma unroll
      for (int e = 0; e < 8; ++e) accO[e][r] *= fs;
      sP[wave][lh * 8 + r][lm]      = f32_bf16(p0);
      sP[wave][lh * 8 + r][16 + lm] = f32_bf16(p1);
    }

    // P back in A-layout (wave-private LDS; same-wave DS ops are in-order)
    Frag16 fp;
    fp.q[0] = *(const uint4*)&sP[wave][lm][lh * 16];
    fp.q[1] = *(const uint4*)&sP[wave][lm][lh * 16 + 8];

    // O += P @ V : 8 output tiles, V fragments batched in groups of 4
    #pragma unroll
    for (int g = 0; g < 2; ++g) {
      Frag16 fv[4];
      #pragma unroll
      for (int j = 0; j < 4; ++j) {
        const unsigned short* vp = Vh + (size_t)((g * 4 + j) * 16 + lm) * S_ + kb + lh * 16;
        fv[j].q[0] = ((const uint4*)vp)[0]; fv[j].q[1] = ((const uint4*)vp)[1];
      }
      #pragma unroll
      for (int j = 0; j < 4; ++j)
        accO[g * 4 + j] = __builtin_amdgcn_wmma_f32_16x16x32_bf16(
            false, fp.m, false, fv[j].m, (short)0, accO[g * 4 + j], false, false);
    }
  }

  #pragma unroll
  for (int e = 0; e < 8; ++e)
    #pragma unroll
    for (int r = 0; r < 8; ++r) {
      int qi = qRow0 + lh * 8 + r;
      float v = accO[e][r] / rowsum[r];
      Am[(size_t)(b * S_ + qi) * E_ + hh * 128 + e * 16 + lm] = v;
    }
}

// ---------------- differential combine + per-head RMS norm -> bf16 ----------------
__global__ __launch_bounds__(128) void k_combine_rms(
    const float* __restrict__ A1, const float* __restrict__ A2,
    const float* __restrict__ lam_p, const float* __restrict__ alnw,
    const float* __restrict__ alnb, float one_minus_li,
    unsigned short* __restrict__ out)
{
  __shared__ float red[128];
  int row = blockIdx.x, hh = blockIdx.y, t = threadIdx.x;
  size_t idx = (size_t)row * E_ + hh * 128 + t;
  float lam = *lam_p;
  float v = A1[idx] - lam * A2[idx];
  red[t] = v * v; __syncthreads();
  for (int o = 64; o > 0; o >>= 1) {
    if (t < o) red[t] += red[t + o];
    __syncthreads();
  }
  float ms = red[0] * (1.f / 128.f);
  float r = v * rsqrtf(ms + 1e-8f) * alnw[t] + alnb[t];
  out[idx] = f32_bf16(r * one_minus_li);
}

// ---------------- gated residual: h = t*sigmoid(1-g) + h ----------------
__global__ __launch_bounds__(256) void k_gate_residual(
    const float* __restrict__ t_in, const float* __restrict__ mods,
    int modsStride, int gOff, float* __restrict__ h)
{
  int row = blockIdx.x;
  int batch = row / S_;
  const float* mrow = mods + (size_t)batch * modsStride + gOff;
  for (int c = threadIdx.x; c < E_; c += 256) {
    float g = mrow[c];
    float gate = 1.f / (1.f + __expf(-(1.f - g)));
    size_t idx = (size_t)row * E_ + c;
    h[idx] = t_in[idx] * gate + h[idx];
  }
}

// ================= host orchestration =================
extern "C" void kernel_launch(void* const* d_in, const int* in_sizes, int n_in,
                              void* d_out, int out_size, void* d_ws, size_t ws_size,
                              hipStream_t stream) {
  (void)in_sizes; (void)n_in; (void)out_size; (void)ws_size;

  const float* x       = (const float*)d_in[0];
  const float* emb     = (const float*)d_in[1];
  const float* l2e_w1  = (const float*)d_in[2];
  const float* l2e_b1  = (const float*)d_in[3];
  const float* l2e_al  = (const float*)d_in[4];
  const float* l2e_be  = (const float*)d_in[5];
  const float* l2e_w2  = (const float*)d_in[6];
  const float* l2e_b2  = (const float*)d_in[7];
  const float* ln1_w   = (const float*)d_in[8];
  const float* ln1_b   = (const float*)d_in[9];
  const float* ln2_w   = (const float*)d_in[10];
  const float* ln2_b   = (const float*)d_in[11];
  const float* qkv_w   = (const float*)d_in[12];
  const float* out_w   = (const float*)d_in[13];
  const float* lq1     = (const float*)d_in[14];
  const float* lk1     = (const float*)d_in[15];
  const float* lq2     = (const float*)d_in[16];
  const float* lk2     = (const float*)d_in[17];
  const float* aln_w   = (const float*)d_in[18];
  const float* aln_b   = (const float*)d_in[19];
  const float* ada_w   = (const float*)d_in[20];
  const float* ada_b   = (const float*)d_in[21];
  const float* ff_w1   = (const float*)d_in[22];
  const float* ff_b1   = (const float*)d_in[23];
  const float* ff_al   = (const float*)d_in[24];
  const float* ff_be   = (const float*)d_in[25];
  const float* ff_w2   = (const float*)d_in[26];
  const float* ff_b2   = (const float*)d_in[27];
  const float* adaf_w  = (const float*)d_in[28];
  const float* adaf_b  = (const float*)d_in[29];
  const float* fin_w   = (const float*)d_in[30];
  const float* fin_b   = (const float*)d_in[31];

  // ---- workspace bump allocator ----
  char* wsb = (char*)d_ws;
  size_t off = 0;
  auto alloc = [&](size_t bytes) -> void* {
    void* p = (void*)(wsb + off);
    off += (bytes + 255) & ~(size_t)255;
    return p;
  };
  unsigned short* xbf   = (unsigned short*)alloc((size_t)M_ * LAT_ * 2);
  unsigned short* w1T   = (unsigned short*)alloc((size_t)E_ * LAT_ * 2);   // [E][LAT]
  unsigned short* w2T   = (unsigned short*)alloc((size_t)E_ * E_ * 2);
  unsigned short* finT  = (unsigned short*)alloc((size_t)LAT_ * E_ * 2);   // [LAT][E]
  unsigned short* wqT   = (unsigned short*)alloc((size_t)3 * E_ * E_ * 2); // [3E][E]
  unsigned short* woT   = (unsigned short*)alloc((size_t)E_ * E_ * 2);
  unsigned short* wf1T  = (unsigned short*)alloc((size_t)E4_ * E_ * 2);    // [E4][E]
  unsigned short* wf2T  = (unsigned short*)alloc((size_t)E_ * E4_ * 2);    // [E][E4]
  float*          hbuf  = (float*)alloc((size_t)M_ * E_ * 4);
  unsigned short* tb    = (unsigned short*)alloc((size_t)M_ * E_ * 2);
  unsigned short* ub    = (unsigned short*)alloc((size_t)M_ * E4_ * 2);
  float*          qkvb  = (float*)alloc((size_t)M_ * 3 * E_ * 4);
  unsigned short* Qb    = (unsigned short*)alloc((size_t)B_ * 2 * H_ * S_ * HD_ * 2);
  unsigned short* Kb    = (unsigned short*)alloc((size_t)B_ * 2 * H_ * S_ * HD_ * 2);
  unsigned short* Vtb   = (unsigned short*)alloc((size_t)B_ * H_ * 128 * S_ * 2);
  float*          a1    = (float*)alloc((size_t)M_ * E_ * 4);
  float*          a2    = (float*)alloc((size_t)M_ * E_ * 4);
  float*          tf    = (float*)alloc((size_t)M_ * E_ * 4);
  float*          mods  = (float*)alloc((size_t)B_ * 6 * E_ * 4);
  float*          modsF = (float*)alloc((size_t)B_ * 2 * E_ * 4);
  float*          lamp  = (float*)alloc(256);

  auto conv = [&](const float* src, unsigned short* dst, size_t n) {
    k_f32_to_bf16<<<(int)((n + 255) / 256), 256, 0, stream>>>(src, dst, (int)n);
  };
  auto transp = [&](const float* W, unsigned short* Wt, int K, int N) {
    dim3 g(N / 32, K / 32);
    k_transpose_bf16<<<g, 256, 0, stream>>>(W, Wt, K, N);
  };
  auto gemm = [&](const unsigned short* A, const unsigned short* Bt, const float* bias,
                  const float* aa, const float* bb, void* C,
                  int M, int N, int K, int act, int obf) {
    dim3 g((N + 255) / 256, M / 32);
    k_gemm_bf16<<<g, 256, 0, stream>>>(A, Bt, bias, aa, bb, C, M, N, K, act, obf);
  };

  // ---- latent-to-embedding ----
  conv(x, xbf, (size_t)M_ * LAT_);
  transp(l2e_w1, w1T, LAT_, E_);
  transp(l2e_w2, w2T, E_, E_);
  transp(fin_w, finT, E_, LAT_);
  gemm(xbf, w1T, l2e_b1, l2e_al, l2e_be, tb, M_, E_, LAT_, 1, 1);   // snake -> bf16
  gemm(tb, w2T, l2e_b2, nullptr, nullptr, hbuf, M_, E_, E_, 0, 0);  // -> f32 h

  // ---- transformer blocks ----
  for (int i = 0; i < NB_; ++i) {
    float li = 0.8f - 0.6f * std::exp(-0.3f * (float)(i + 1));

    k_emb_mm<<<(6 * E_) / 256, 256, 0, stream>>>(emb, ada_w + (size_t)i * E_ * 6 * E_,
                                                 ada_b + (size_t)i * 6 * E_, mods, 6 * E_);
    transp(qkv_w + (size_t)i * E_ * 3 * E_, wqT, E_, 3 * E_);
    transp(out_w + (size_t)i * E_ * E_, woT, E_, E_);
    transp(ff_w1 + (size_t)i * E_ * E4_, wf1T, E_, E4_);
    transp(ff_w2 + (size_t)i * E4_ * E_, wf2T, E4_, E_);

    // attention branch
    k_mod_ln<<<M_, 256, 0, stream>>>(hbuf, ln1_w + (size_t)i * E_, ln1_b + (size_t)i * E_,
                                     mods, 6 * E_, 0, E_, 1e-5f, tb);
    gemm(tb, wqT, nullptr, nullptr, nullptr, qkvb, M_, 3 * E_, E_, 0, 0);
    k_rotary_pack<<<M_, 256, 0, stream>>>(qkvb, Qb, Kb, Vtb);
    k_lambda<<<1, 64, 0, stream>>>(lq1 + (size_t)i * HD_, lk1 + (size_t)i * HD_,
                                   lq2 + (size_t)i * HD_, lk2 + (size_t)i * HD_, li, lamp);
    {
      dim3 fg(S_ / 64, H_, B_ * 2);
      k_flash_attn<<<fg, 128, 0, stream>>>(Qb, Kb, Vtb, a1, a2);
    }
    {
      dim3 cg(M_, H_);
      k_combine_rms<<<cg, 128, 0, stream>>>(a1, a2, lamp, aln_w + (size_t)i * 128,
                                            aln_b + (size_t)i * 128, 1.f - li, tb);
    }
    gemm(tb, woT, nullptr, nullptr, nullptr, tf, M_, E_, E_, 0, 0);
    k_gate_residual<<<M_, 256, 0, stream>>>(tf, mods, 6 * E_, 4 * E_, hbuf);

    // MLP branch
    k_mod_ln<<<M_, 256, 0, stream>>>(hbuf, ln2_w + (size_t)i * E_, ln2_b + (size_t)i * E_,
                                     mods, 6 * E_, 2 * E_, 3 * E_, 1e-5f, tb);
    gemm(tb, wf1T, ff_b1 + (size_t)i * E4_, ff_al + (size_t)i * E4_,
         ff_be + (size_t)i * E4_, ub, M_, E4_, E_, 1, 1);            // snake -> bf16
    gemm(ub, wf2T, ff_b2 + (size_t)i * E_, nullptr, nullptr, tf, M_, E_, E4_, 0, 0);
    k_gate_residual<<<M_, 256, 0, stream>>>(tf, mods, 6 * E_, 5 * E_, hbuf);
  }

  // ---- final modulated LN (no affine) + projection ----
  k_emb_mm<<<(2 * E_) / 256, 256, 0, stream>>>(emb, adaf_w, adaf_b, modsF, 2 * E_);
  k_mod_ln<<<M_, 256, 0, stream>>>(hbuf, nullptr, nullptr, modsF, 2 * E_, 0, E_, 1e-6f, tb);
  gemm(tb, finT, fin_b, nullptr, nullptr, d_out, M_, LAT_, E_, 0, 0);
}
